// SwinTransformerNet_46454366273923
// MI455X (gfx1250) — compile-verified, matrix operand order checked by reading, not run on hardware
//
#include <hip/hip_runtime.h>
#include <hip/hip_bf16.h>
#include <cstdint>
#include <cstddef>

typedef __attribute__((ext_vector_type(16))) _Float16 v16h;
typedef __attribute__((ext_vector_type(8)))  float    v8f;

// ---------------------------------------------------------------------------
// WMMA fragment helpers (16x16x32 f16 -> f32), per CDNA5 ISA VGPR layouts.
// A fragment (16 rows x 32 k), source row-major [row][ld]:
//   lane L: row = L&15, k-base = (L>>4)*8; element pairs k = g*16 + kb + 2j
// B fragment (32 k x 16 cols), source row-major [k][ld] (cols striped in lanes)
// ---------------------------------------------------------------------------
__device__ __forceinline__ v8f vzero8() {
  v8f z;
  #pragma unroll
  for (int i = 0; i < 8; ++i) z[i] = 0.0f;
  return z;
}

__device__ __forceinline__ v16h load_frag_row(const _Float16* __restrict__ s, int ld) {
  const int lane = threadIdx.x & 31;
  const _Float16* row = s + (size_t)(lane & 15) * ld + ((lane >> 4) << 3);
  v16h f;
  #pragma unroll
  for (int g = 0; g < 2; ++g) {
    #pragma unroll
    for (int j = 0; j < 4; ++j) {
      f[g * 8 + 2 * j]     = row[g * 16 + 2 * j];
      f[g * 8 + 2 * j + 1] = row[g * 16 + 2 * j + 1];
    }
  }
  return f;
}

__device__ __forceinline__ v16h load_frag_col(const _Float16* __restrict__ s, int ld) {
  const int lane = threadIdx.x & 31;
  const int n  = lane & 15;
  const int kb = (lane >> 4) << 3;
  v16h f;
  #pragma unroll
  for (int g = 0; g < 2; ++g) {
    #pragma unroll
    for (int j = 0; j < 4; ++j) {
      const int k = g * 16 + kb + 2 * j;
      f[g * 8 + 2 * j]     = s[(size_t)k * ld + n];
      f[g * 8 + 2 * j + 1] = s[(size_t)(k + 1) * ld + n];
    }
  }
  return f;
}

__device__ __forceinline__ float gelu_tanh(float x) {
  const float c = 0.7978845608028654f;
  return 0.5f * x * (1.0f + tanhf(c * (x + 0.044715f * x * x * x)));
}

// ---------------------------------------------------------------------------
// Generic WMMA GEMM: C[M,N] = A[M,K](f16) * B[K,N](f32, cvt->f16) + epilogue
// EPI: 0=none  1=+bias  2=+bias,GELU  3=+bias,+residual
// OUTF16: write f16 (feeds next GEMM) or f32 (residual stream)
// block = 128 threads (4 waves), tile 64x64, K step 32 (one WMMA per step)
// ---------------------------------------------------------------------------
#define BM 64
#define BN 64
#define BK 32

template <int EPI, bool OUTF16>
__global__ __launch_bounds__(128) void gemm_wmma(
    const _Float16* __restrict__ A, const float* __restrict__ B,
    const float* __restrict__ bias, const float* __restrict__ res,
    void* __restrict__ Out, int M, int N, int K) {
  __shared__ _Float16 As[BM][BK + 8];
  __shared__ _Float16 Bs[BK][BN + 8];

  const int tid = threadIdx.x;
  const int wid = tid >> 5;
  const int m0 = blockIdx.y * BM;
  const int n0 = blockIdx.x * BN;
  const int wm = (wid >> 1) * 32;  // wave sub-tile origin inside block tile
  const int wn = (wid & 1) * 32;

  v8f acc[2][2];
  #pragma unroll
  for (int a = 0; a < 2; ++a)
    #pragma unroll
    for (int b = 0; b < 2; ++b) acc[a][b] = vzero8();

  for (int k0 = 0; k0 < K; k0 += BK) {
    // stage A tile (f16, zero-pad out of range)
    #pragma unroll 4
    for (int e = tid; e < BM * BK; e += 128) {
      const int r = e >> 5, c = e & 31;
      const int gm = m0 + r, gk = k0 + c;
      As[r][c] = (gm < M && gk < K) ? A[(size_t)gm * K + gk] : (_Float16)0.0f;
    }
    // stage B tile (f32 -> f16)
    #pragma unroll 4
    for (int e = tid; e < BK * BN; e += 128) {
      const int r = e >> 6, c = e & 63;
      const int gk = k0 + r, gn = n0 + c;
      Bs[r][c] = (gk < K && gn < N) ? (_Float16)B[(size_t)gk * N + gn] : (_Float16)0.0f;
    }
    if (k0 + BK < K) {  // hint next tiles toward L2
      __builtin_prefetch(&A[(size_t)(m0 + (tid >> 1)) * K + k0 + BK], 0, 1);
      __builtin_prefetch(&B[(size_t)(k0 + BK + (tid & 31)) * N + n0], 0, 1);
    }
    __syncthreads();

    #pragma unroll
    for (int tm = 0; tm < 2; ++tm) {
      const v16h af = load_frag_row(&As[wm + tm * 16][0], BK + 8);
      #pragma unroll
      for (int tn = 0; tn < 2; ++tn) {
        const v16h bf = load_frag_col(&Bs[0][wn + tn * 16], BN + 8);
        acc[tm][tn] = __builtin_amdgcn_wmma_f32_16x16x32_f16(
            false, af, false, bf, (short)0, acc[tm][tn], false, false);
      }
    }
    __syncthreads();
  }

  // epilogue: C layout -> lane holds n=(lane&15), rows r + 8*(lane>>4)
  const int lane = tid & 31;
  const int ln = lane & 15, lh = lane >> 4;
  #pragma unroll
  for (int tm = 0; tm < 2; ++tm) {
    #pragma unroll
    for (int tn = 0; tn < 2; ++tn) {
      #pragma unroll
      for (int r = 0; r < 8; ++r) {
        const int gm = m0 + wm + tm * 16 + r + 8 * lh;
        const int gn = n0 + wn + tn * 16 + ln;
        if (gm < M && gn < N) {
          float v = acc[tm][tn][r];
          if (EPI >= 1) v += bias[gn];
          if (EPI == 2) v = gelu_tanh(v);
          if (EPI == 3) v += res[(size_t)gm * N + gn];
          if (OUTF16)
            ((_Float16*)Out)[(size_t)gm * N + gn] = (_Float16)v;
          else
            ((float*)Out)[(size_t)gm * N + gn] = v;
        }
      }
    }
  }
}

// ---------------------------------------------------------------------------
// Fused window attention: one workgroup (4 waves) per (batch, window, head).
// Head dim = 32 everywhere -> QK^T is a single-K-step WMMA chain.
// Roll + window partition/reverse folded into the index math. Relative-position
// bias gathered on the fly; shift mask computed analytically.
// ---------------------------------------------------------------------------
__device__ __forceinline__ int sw_region(int H, int c) {
  return (c < H - 7) ? 0 : ((c < H - 3) ? 1 : 2);
}

__global__ __launch_bounds__(128) void attn_win(
    const _Float16* __restrict__ qkv, _Float16* __restrict__ out,
    const float* __restrict__ relb, int Bn, int H, int nH, int shift) {
  const int nWx = H / 7;
  const int nWin = nWx * nWx;
  int idx = blockIdx.x;
  const int h = idx % nH;   idx /= nH;
  const int w = idx % nWin; idx /= nWin;
  const int b = idx;
  const int wy = w / nWx, wx = w % nWx;
  const int C = nH * 32;
  const int tid = threadIdx.x;
  const int wid = tid >> 5;
  const int lane = tid & 31;

  __shared__ _Float16 Qs[64][40];
  __shared__ _Float16 Ks[64][40];
  __shared__ _Float16 Vs[64][40];
  __shared__ float    Sc[4][16][65];
  __shared__ _Float16 Ps[4][16][72];

  auto token = [&](int r) -> size_t {
    const int ry = r / 7, rx = r % 7;
    int oy = wy * 7 + ry + shift; if (oy >= H) oy -= H;
    int ox = wx * 7 + rx + shift; if (ox >= H) ox -= H;
    return ((size_t)b * H + oy) * H + ox;
  };

  const float qscale = 0.17677669529663687f;  // 1/sqrt(32)
  #pragma unroll 4
  for (int e = tid; e < 64 * 32; e += 128) {
    const int r = e >> 5, d = e & 31;
    _Float16 qv = (_Float16)0.0f, kv = (_Float16)0.0f, vv = (_Float16)0.0f;
    if (r < 49) {
      const size_t base = token(r) * (size_t)(3 * C) + (size_t)h * 32 + d;
      qv = (_Float16)((float)qkv[base] * qscale);
      kv = qkv[base + C];
      vv = qkv[base + 2 * C];
    }
    Qs[r][d] = qv; Ks[r][d] = kv; Vs[r][d] = vv;
  }
  __syncthreads();

  // --- scores = Q * K^T : wave `wid` owns query rows [16*wid, 16*wid+16) ---
  {
    const v16h aq = load_frag_row(&Qs[wid * 16][0], 40);
    const int ln = lane & 15, lh = lane >> 4;
    #pragma unroll
    for (int nt = 0; nt < 4; ++nt) {
      // B(d,kcol) = K[kcol][d]: row-major K rows == A-style fragment
      const v16h bk = load_frag_row(&Ks[nt * 16][0], 40);
      v8f c = vzero8();
      c = __builtin_amdgcn_wmma_f32_16x16x32_f16(false, aq, false, bk,
                                                 (short)0, c, false, false);
      #pragma unroll
      for (int r = 0; r < 8; ++r) Sc[wid][r + 8 * lh][nt * 16 + ln] = c[r];
    }
  }
  __syncthreads();

  // --- softmax rows (bias + shift-mask fused), lanes 0..15 of each wave ---
  if (lane < 16) {
    const int q = wid * 16 + lane;
    float*     srow = &Sc[wid][lane][0];
    _Float16*  prow = &Ps[wid][lane][0];
    if (q < 49) {
      const int qy = q / 7, qx = q % 7;
      int lq = 0;
      if (shift > 0) lq = sw_region(H, wy * 7 + qy) * 3 + sw_region(H, wx * 7 + qx);
      float mx = -1e30f;
      for (int k = 0; k < 49; ++k) {
        const int ky = k / 7, kx = k % 7;
        float v = srow[k] + relb[((qy - ky + 6) * 13 + (qx - kx + 6)) * nH + h];
        if (shift > 0) {
          const int lk = sw_region(H, wy * 7 + ky) * 3 + sw_region(H, wx * 7 + kx);
          if (lk != lq) v -= 100.0f;
        }
        srow[k] = v;
        mx = fmaxf(mx, v);
      }
      float sum = 0.0f;
      for (int k = 0; k < 49; ++k) { const float e = __expf(srow[k] - mx); srow[k] = e; sum += e; }
      const float inv = 1.0f / sum;
      for (int k = 0; k < 49; ++k) prow[k] = (_Float16)(srow[k] * inv);
      for (int k = 49; k < 64; ++k) prow[k] = (_Float16)0.0f;
    } else {
      for (int k = 0; k < 64; ++k) prow[k] = (_Float16)0.0f;
    }
  }
  __syncthreads();

  // --- out = P * V : K dim 64 (2 steps), N dim 32 (2 tiles) ---
  v8f o[2] = {vzero8(), vzero8()};
  #pragma unroll
  for (int kc = 0; kc < 2; ++kc) {
    const v16h ap = load_frag_row(&Ps[wid][0][kc * 32], 72);
    #pragma unroll
    for (int nt = 0; nt < 2; ++nt) {
      const v16h bv = load_frag_col(&Vs[kc * 32][nt * 16], 40);
      o[nt] = __builtin_amdgcn_wmma_f32_16x16x32_f16(false, ap, false, bv,
                                                     (short)0, o[nt], false, false);
    }
  }
  const int ln = lane & 15, lh = lane >> 4;
  #pragma unroll
  for (int nt = 0; nt < 2; ++nt) {
    #pragma unroll
    for (int r = 0; r < 8; ++r) {
      const int q = wid * 16 + r + 8 * lh;
      if (q < 49) {
        const size_t t = token(q);  // same map reverses the roll
        out[t * C + (size_t)h * 32 + nt * 16 + ln] = (_Float16)o[nt][r];
      }
    }
  }
}

// ---------------------------------------------------------------------------
// LayerNorm: one wave per row, shfl_xor tree reduction (wave32)
// ---------------------------------------------------------------------------
template <bool OUTF16>
__global__ __launch_bounds__(256) void layernorm_k(
    const float* __restrict__ x, const float* __restrict__ g,
    const float* __restrict__ be, void* __restrict__ out, int M, int C) {
  const int row = blockIdx.x * 8 + (threadIdx.x >> 5);
  if (row >= M) return;
  const int lane = threadIdx.x & 31;
  const float* xr = x + (size_t)row * C;
  float s = 0.0f, s2 = 0.0f;
  for (int c = lane; c < C; c += 32) { const float v = xr[c]; s += v; s2 += v * v; }
  #pragma unroll
  for (int o = 16; o > 0; o >>= 1) { s += __shfl_xor(s, o, 32); s2 += __shfl_xor(s2, o, 32); }
  const float mean = s / C;
  const float inv = rsqrtf(s2 / C - mean * mean + 1e-5f);
  for (int c = lane; c < C; c += 32) {
    const float v = (xr[c] - mean) * inv * g[c] + be[c];
    if (OUTF16) ((_Float16*)out)[(size_t)row * C + c] = (_Float16)v;
    else        ((float*)out)[(size_t)row * C + c] = v;
  }
}

// ---------------------------------------------------------------------------
// Patch merging: gather 2x2 neighborhood (concat order 00,10,01,11), LN(4C),
// emit f16 rows for the reduction GEMM. One wave per output token.
// ---------------------------------------------------------------------------
__global__ __launch_bounds__(256) void merge_ln_k(
    const float* __restrict__ x, const float* __restrict__ g,
    const float* __restrict__ be, _Float16* __restrict__ out,
    int Bn, int H, int C) {
  const int H2 = H / 2;
  const int M2 = Bn * H2 * H2;
  const int C4 = 4 * C;
  const int row = blockIdx.x * 8 + (threadIdx.x >> 5);
  if (row >= M2) return;
  const int lane = threadIdx.x & 31;
  const int b = row / (H2 * H2);
  const int pos = row % (H2 * H2);
  const int y2 = pos / H2, x2 = pos % H2;
  const int dy[4] = {0, 1, 0, 1};
  const int dx[4] = {0, 0, 1, 1};
  auto gather = [&](int c) -> float {
    const int ch = c / C, cc = c % C;
    return x[(((size_t)b * H + (2 * y2 + dy[ch])) * H + (2 * x2 + dx[ch])) * C + cc];
  };
  float s = 0.0f, s2 = 0.0f;
  for (int c = lane; c < C4; c += 32) { const float v = gather(c); s += v; s2 += v * v; }
  #pragma unroll
  for (int o = 16; o > 0; o >>= 1) { s += __shfl_xor(s, o, 32); s2 += __shfl_xor(s2, o, 32); }
  const float mean = s / C4;
  const float inv = rsqrtf(s2 / C4 - mean * mean + 1e-5f);
  for (int c = lane; c < C4; c += 32)
    out[(size_t)row * C4 + c] = (_Float16)((gather(c) - mean) * inv * g[c] + be[c]);
}

// ---------------------------------------------------------------------------
// Patch embed helpers: im2col (f16) + weight repack OIHW -> [K=48][N=96] f32
// ---------------------------------------------------------------------------
__global__ void im2col_k(const float* __restrict__ x, _Float16* __restrict__ out, int total) {
  const int e = blockIdx.x * 256 + threadIdx.x;
  if (e >= total) return;
  const int m = e / 48, k = e % 48;
  const int b = m / 3136, pos = m % 3136;
  const int y = pos / 56, xx = pos % 56;
  const int i = k >> 4, kh = (k >> 2) & 3, kw = k & 3;
  out[e] = (_Float16)x[(((size_t)b * 3 + i) * 224 + (y * 4 + kh)) * 224 + (xx * 4 + kw)];
}

__global__ void repack_pw_k(const float* __restrict__ w, float* __restrict__ out) {
  const int e = blockIdx.x * 256 + threadIdx.x;
  if (e >= 48 * 96) return;
  const int k = e / 96, n = e % 96;
  const int i = k >> 4, kh = (k >> 2) & 3, kw = k & 3;
  out[e] = w[(((size_t)n * 3 + i) * 4 + kh) * 4 + kw];
}

// ---------------------------------------------------------------------------
// Host orchestration
// ---------------------------------------------------------------------------
extern "C" void kernel_launch(void* const* d_in, const int* in_sizes, int n_in,
                              void* d_out, int out_size, void* d_ws, size_t ws_size,
                              hipStream_t stream) {
  (void)in_sizes; (void)n_in; (void)out_size; (void)ws_size;

  static const int DEP[4]   = {2, 2, 6, 2};
  static const int NHEAD[4] = {3, 6, 12, 24};
  static const int CDIM[4]  = {96, 192, 384, 768};
  const int Bn = 32;

  int p = 0;
  const float* x_in     = (const float*)d_in[p++];
  const float* patch_w  = (const float*)d_in[p++];
  const float* patch_b  = (const float*)d_in[p++];
  const float* patch_g  = (const float*)d_in[p++];
  const float* patch_be = (const float*)d_in[p++];

  struct Blk {
    const float *n1g, *n1b, *qkvw, *qkvb, *relb, *projw, *projb;
    const float *n2g, *n2b, *fc1w, *fc1b, *fc2w, *fc2b;
  };
  Blk blk[4][6];
  const float *dsg[3], *dsb[3], *dsw[3];
  for (int i = 0; i < 4; ++i) {
    for (int j = 0; j < DEP[i]; ++j) {
      Blk& k = blk[i][j];
      k.n1g   = (const float*)d_in[p++];
      k.n1b   = (const float*)d_in[p++];
      k.qkvw  = (const float*)d_in[p++];
      k.qkvb  = (const float*)d_in[p++];
      k.relb  = (const float*)d_in[p++];
      k.projw = (const float*)d_in[p++];
      k.projb = (const float*)d_in[p++];
      k.n2g   = (const float*)d_in[p++];
      k.n2b   = (const float*)d_in[p++];
      k.fc1w  = (const float*)d_in[p++];
      k.fc1b  = (const float*)d_in[p++];
      k.fc2w  = (const float*)d_in[p++];
      k.fc2b  = (const float*)d_in[p++];
    }
    if (i < 3) {
      dsg[i] = (const float*)d_in[p++];
      dsb[i] = (const float*)d_in[p++];
      dsw[i] = (const float*)d_in[p++];
    }
  }

  // ---- workspace carve -------------------------------------------------
  char* wp = (char*)d_ws;
  auto carve = [&](size_t bytes) {
    char* r = wp;
    wp += (bytes + 255) & ~(size_t)255;
    return r;
  };
  const size_t M0 = (size_t)Bn * 56 * 56;  // 100352 tokens, stage 0
  float*     X     = (float*)carve(M0 * 96 * sizeof(float));
  float*     X2    = (float*)carve(M0 * 96 * sizeof(float));
  _Float16*  LN16  = (_Float16*)carve(M0 * 96 * sizeof(_Float16));
  _Float16*  QKV16 = (_Float16*)carve(M0 * 288 * sizeof(_Float16));
  _Float16*  ATT16 = (_Float16*)carve(M0 * 96 * sizeof(_Float16));
  _Float16*  H16   = (_Float16*)carve(M0 * 384 * sizeof(_Float16));
  float*     PW    = (float*)carve(48 * 96 * sizeof(float));
  _Float16*  COL   = (_Float16*)carve(M0 * 48 * sizeof(_Float16));

  auto gemm = [&](const _Float16* A, const float* B, const float* bias,
                  const float* res, void* Out, int M, int N, int K,
                  int epi, bool outf16) {
    dim3 g((N + BN - 1) / BN, (M + BM - 1) / BM), t(128);
    if (epi == 0 && !outf16)
      gemm_wmma<0, false><<<g, t, 0, stream>>>(A, B, bias, res, Out, M, N, K);
    else if (epi == 1 && !outf16)
      gemm_wmma<1, false><<<g, t, 0, stream>>>(A, B, bias, res, Out, M, N, K);
    else if (epi == 1 && outf16)
      gemm_wmma<1, true><<<g, t, 0, stream>>>(A, B, bias, res, Out, M, N, K);
    else if (epi == 2 && outf16)
      gemm_wmma<2, true><<<g, t, 0, stream>>>(A, B, bias, res, Out, M, N, K);
    else
      gemm_wmma<3, false><<<g, t, 0, stream>>>(A, B, bias, res, Out, M, N, K);
  };

  // ---- patch embed: im2col GEMM + LayerNorm ----------------------------
  {
    const int total = (int)(M0 * 48);
    im2col_k<<<(total + 255) / 256, 256, 0, stream>>>(x_in, COL, total);
    repack_pw_k<<<(48 * 96 + 255) / 256, 256, 0, stream>>>(patch_w, PW);
    gemm(COL, PW, patch_b, nullptr, X2, (int)M0, 96, 48, /*epi=*/1, /*f16=*/false);
    layernorm_k<false><<<((int)M0 + 7) / 8, 256, 0, stream>>>(X2, patch_g, patch_be, X, (int)M0, 96);
  }

  // ---- stages ----------------------------------------------------------
  int H = 56;
  size_t out_off = 0;
  float* cur = X;
  float* alt = X2;
  for (int i = 0; i < 4; ++i) {
    const int C = CDIM[i], nH = NHEAD[i];
    const int M = Bn * H * H;
    const int nWin = (H / 7) * (H / 7);
    for (int j = 0; j < DEP[i]; ++j) {
      const Blk& b = blk[i][j];
      const int shift = (H > 7 && (j & 1)) ? 3 : 0;
      // attn branch
      layernorm_k<true><<<(M + 7) / 8, 256, 0, stream>>>(cur, b.n1g, b.n1b, LN16, M, C);
      gemm(LN16, b.qkvw, b.qkvb, nullptr, QKV16, M, 3 * C, C, 1, true);
      attn_win<<<Bn * nWin * nH, 128, 0, stream>>>(QKV16, ATT16, b.relb, Bn, H, nH, shift);
      gemm(ATT16, b.projw, b.projb, cur, cur, M, C, C, 3, false);
      // mlp branch
      layernorm_k<true><<<(M + 7) / 8, 256, 0, stream>>>(cur, b.n2g, b.n2b, LN16, M, C);
      gemm(LN16, b.fc1w, b.fc1b, nullptr, H16, M, 4 * C, C, 2, true);
      gemm(H16, b.fc2w, b.fc2b, cur, cur, M, C, 4 * C, 3, false);
    }
    size_t out_elems;
    if (i < 3) {
      const int M2 = M / 4;
      merge_ln_k<<<(M2 + 7) / 8, 256, 0, stream>>>(cur, dsg[i], dsb[i], H16, Bn, H, C);
      gemm(H16, dsw[i], nullptr, nullptr, alt, M2, 2 * C, 4 * C, 0, false);
      float* t = cur; cur = alt; alt = t;
      H /= 2;
      out_elems = (size_t)M2 * (2 * C);
    } else {
      out_elems = (size_t)M * C;
    }
    (void)hipMemcpyAsync((float*)d_out + out_off, cur, out_elems * sizeof(float),
                         hipMemcpyDeviceToDevice, stream);
    out_off += out_elems;
  }
}
